// RWKV_7_32031866093783
// MI455X (gfx1250) — compile-verified, matrix operand order checked by reading, not run on hardware
//
#include <hip/hip_runtime.h>
#include <hip/hip_bf16.h>
#include <math.h>

// ---------------- model constants ----------------
#define LL 2
#define TT 1024
#define CC 2048
#define HH 32
#define NN 64
#define VV 50304
#define FHH 8192
#define DWW 64
#define DAA 64
#define DVV 32
#define DGG 128

typedef __attribute__((ext_vector_type(16))) _Float16 v16h;
typedef __attribute__((ext_vector_type(8)))  float    v8f;
typedef __attribute__((ext_vector_type(4)))  _Float16 h4;

// =====================================================================
// WMMA NT GEMM:  out[m,n] = sum_k A[m,k] * W[n,k]    (A:[M,K], W:[N,K])
// f32 -> f16 convert on LDS stage, f32 accumulate via v_wmma_f32_16x16x32_f16.
// Block = 256 threads = 8 waves; block tile 32(M) x 128(N); K panel 64.
// Waves: 2 along M x 4 along N; each wave: two 16x16 accumulators (share af).
// Staging gathers all 10 b128 loads into registers first so they overlap,
// then converts/stores to LDS (drained with partial loadcnt waits).
// Requires M%32==0, N%128==0, K%64==0 (true for all call sites here).
// =====================================================================
template <bool ACC, bool RELUSQ>
__global__ __launch_bounds__(256)
void gemm_nt_wmma(const float* __restrict__ A, const float* __restrict__ W,
                  float* __restrict__ out, int M, int N, int K) {
  __shared__ _Float16 As[32][72];    // 64 + 8 pad halves; rows 144B (16B aligned)
  __shared__ _Float16 Bs[128][72];
  const int tid  = threadIdx.x;
  const int bm   = blockIdx.x * 32;
  const int bn   = blockIdx.y * 128;
  const int wave = tid >> 5;
  const int lane = tid & 31;
  const int wm   = (wave & 1) * 16;   // 2 waves along M
  const int wn   = (wave >> 1) * 32;  // 4 waves along N, 32 cols each

  // per-thread staging coordinates (k-invariant)
  int ra[2], ca[2], rb[8], cb[8];
#pragma unroll
  for (int i = 0; i < 2; ++i) { int e = tid + i * 256; ra[i] = e >> 4; ca[i] = (e & 15) * 4; }
#pragma unroll
  for (int i = 0; i < 8; ++i) { int e = tid + i * 256; rb[i] = e >> 4; cb[i] = (e & 15) * 4; }

  v8f acc0 = {}, acc1 = {};

  for (int k0 = 0; k0 < K; k0 += 64) {
    // (a) gather: issue all 10 independent global_load_b128 first
    float4 fa[2], fb[8];
#pragma unroll
    for (int i = 0; i < 2; ++i)
      fa[i] = *(const float4*)(A + (size_t)(bm + ra[i]) * K + k0 + ca[i]);
#pragma unroll
    for (int i = 0; i < 8; ++i)
      fb[i] = *(const float4*)(W + (size_t)(bn + rb[i]) * K + k0 + cb[i]);
    // prefetch next panel (global_prefetch_b8)
    if (k0 + 64 < K) {
      __builtin_prefetch(&A[(size_t)(bm + (tid >> 3)) * K + k0 + 64], 0, 3);
      __builtin_prefetch(&W[(size_t)(bn + (tid >> 1)) * K + k0 + 64], 0, 3);
    }
    // (b) convert + LDS store
#pragma unroll
    for (int i = 0; i < 2; ++i) {
      h4 hv = {(_Float16)fa[i].x, (_Float16)fa[i].y, (_Float16)fa[i].z, (_Float16)fa[i].w};
      *(h4*)(&As[ra[i]][ca[i]]) = hv;   // ds_store_b64
    }
#pragma unroll
    for (int i = 0; i < 8; ++i) {
      h4 hv = {(_Float16)fb[i].x, (_Float16)fb[i].y, (_Float16)fb[i].z, (_Float16)fb[i].w};
      *(h4*)(&Bs[rb[i]][cb[i]]) = hv;
    }
    __syncthreads();

    // Fragment layout (ISA 7.12.2, 16-bit A 16x32): lane<16 -> K base 0, lane>=16 -> K base 8
    const int mrow = wm + (lane & 15);
    const int kb   = (lane >> 4) * 8;
#pragma unroll
    for (int ks = 0; ks < 64; ks += 32) {
      v16h af;
#pragma unroll
      for (int p = 0; p < 8; ++p) {
        int kk0 = ks + kb + ((p & 3) << 1) + ((p >> 2) << 4);
        af[2 * p]     = As[mrow][kk0];
        af[2 * p + 1] = As[mrow][kk0 + 1];
      }
#pragma unroll
      for (int ns = 0; ns < 2; ++ns) {
        const int ncol = wn + ns * 16 + (lane & 15);
        v16h bf;
#pragma unroll
        for (int p = 0; p < 8; ++p) {
          int kk0 = ks + kb + ((p & 3) << 1) + ((p >> 2) << 4);
          bf[2 * p]     = Bs[ncol][kk0];
          bf[2 * p + 1] = Bs[ncol][kk0 + 1];
        }
        if (ns == 0)
          acc0 = __builtin_amdgcn_wmma_f32_16x16x32_f16(false, af, false, bf,
                                                        (short)0, acc0, false, false);
        else
          acc1 = __builtin_amdgcn_wmma_f32_16x16x32_f16(false, af, false, bf,
                                                        (short)0, acc1, false, false);
      }
    }
    __syncthreads();
  }

  // C/D layout: VGPR j -> M = j + 8*(lane>=16), N = lane&15
  const int mb = (lane >> 4) * 8;
  const int nl = lane & 15;
#pragma unroll
  for (int j = 0; j < 8; ++j) {
    int m = bm + wm + mb + j;
    float s0 = acc0[j], s1 = acc1[j];
    if (RELUSQ) {
      s0 = (s0 > 0.f) ? s0 * s0 : 0.f;
      s1 = (s1 > 0.f) ? s1 * s1 : 0.f;
    }
    size_t o0 = (size_t)m * N + bn + wn + nl;
    size_t o1 = o0 + 16;
    out[o0] = ACC ? (out[o0] + s0) : s0;
    out[o1] = ACC ? (out[o1] + s1) : s1;
  }
}

// =====================================================================
// Embedding gather
// =====================================================================
__global__ void embed_kernel(const int* __restrict__ idx,
                             const float* __restrict__ emb,
                             float* __restrict__ x) {
  int i = blockIdx.x * blockDim.x + threadIdx.x;
  if (i >= TT * CC) return;
  int t = i / CC, c = i % CC;
  x[i] = emb[(size_t)idx[t] * CC + c];
}

// =====================================================================
// LayerNorm over C, one block per row (256 threads, LDS tree reduce)
// =====================================================================
__global__ __launch_bounds__(256)
void layernorm_kernel(const float* __restrict__ in, const float* __restrict__ w,
                      const float* __restrict__ b, float* __restrict__ out,
                      float eps) {
  const int t = blockIdx.x;
  const float* row = in + (size_t)t * CC;
  __shared__ float red[256];
  float s = 0.f;
  for (int c = threadIdx.x; c < CC; c += 256) s += row[c];
  red[threadIdx.x] = s; __syncthreads();
  for (int st = 128; st; st >>= 1) {
    if (threadIdx.x < st) red[threadIdx.x] += red[threadIdx.x + st];
    __syncthreads();
  }
  float mean = red[0] / CC; __syncthreads();
  float v = 0.f;
  for (int c = threadIdx.x; c < CC; c += 256) { float d = row[c] - mean; v += d * d; }
  red[threadIdx.x] = v; __syncthreads();
  for (int st = 128; st; st >>= 1) {
    if (threadIdx.x < st) red[threadIdx.x] += red[threadIdx.x + st];
    __syncthreads();
  }
  float rstd = rsqrtf(red[0] / CC + eps);
  for (int c = threadIdx.x; c < CC; c += 256)
    out[(size_t)t * CC + c] = (row[c] - mean) * rstd * w[c] + b[c];
}

// =====================================================================
// Token-shift mix: out = x + (prev - x) * mixv   (prev row 0 from shift state)
// =====================================================================
__global__ void mix_kernel(const float* __restrict__ x,
                           const float* __restrict__ shiftrow,
                           const float* __restrict__ mixv,
                           float* __restrict__ out) {
  int i = blockIdx.x * blockDim.x + threadIdx.x;
  if (i >= TT * CC) return;
  int t = i / CC, c = i % CC;
  float cur = x[i];
  float prev = (t == 0) ? shiftrow[c] : x[i - CC];
  out[i] = cur + (prev - cur) * mixv[c];
}

// =====================================================================
// Small NN GEMM (LoRA down-proj): out[t,n] = act( sum_k A[t,k]*W[k,n] )
// act: 0 none, 1 tanh, 2 sigmoid
// =====================================================================
__global__ void small_gemm_nn(const float* __restrict__ A,
                              const float* __restrict__ W,
                              float* __restrict__ out, int K, int Nn, int act) {
  int i = blockIdx.x * blockDim.x + threadIdx.x;
  if (i >= TT * Nn) return;
  int t = i / Nn, n = i % Nn;
  float s = 0.f;
  for (int k = 0; k < K; ++k) s += A[(size_t)t * K + k] * W[(size_t)k * Nn + n];
  if (act == 1) s = tanhf(s);
  else if (act == 2) s = 1.f / (1.f + __expf(-s));
  out[i] = s;
}

// LoRA up-proj with fused epilogue. mode: 0 plain, 1 sigmoid(s+bias),
// 2 decay: w=s+bias; w=-softplus(-w)-0.5; out=-exp(w)  (== wlog)
__global__ void lora_out_kernel(const float* __restrict__ Tm,
                                const float* __restrict__ W2,
                                const float* __restrict__ bias,
                                float* __restrict__ out, int D, int mode) {
  int i = blockIdx.x * blockDim.x + threadIdx.x;
  if (i >= TT * CC) return;
  int t = i / CC, c = i % CC;
  float s = 0.f;
  for (int d = 0; d < D; ++d) s += Tm[(size_t)t * D + d] * W2[(size_t)d * CC + c];
  if (mode == 1) {
    s = 1.f / (1.f + __expf(-(s + bias[c])));
  } else if (mode == 2) {
    float w = s + bias[c];
    w = -log1pf(__expf(-w)) - 0.5f;
    s = -__expf(w);
  }
  out[i] = s;
}

// v = v + (v_first - v) * sig
__global__ void vblend_kernel(float* __restrict__ v,
                              const float* __restrict__ vfirst,
                              const float* __restrict__ sig) {
  int i = blockIdx.x * blockDim.x + threadIdx.x;
  if (i >= TT * CC) return;
  v[i] = v[i] + (vfirst[i] - v[i]) * sig[i];
}

// kk = normalize(k * k_k) per head (thread per (t,h))
__global__ void kk_norm_kernel(const float* __restrict__ k,
                               const float* __restrict__ k_k,
                               float* __restrict__ kk) {
  int i = blockIdx.x * blockDim.x + threadIdx.x;
  if (i >= TT * HH) return;
  int t = i / HH, h = i % HH;
  const float* kr = k + (size_t)t * CC + h * NN;
  const float* kkw = k_k + h * NN;
  float ss = 0.f;
  for (int n = 0; n < NN; ++n) { float e = kr[n] * kkw[n]; ss += e * e; }
  float inv = 1.f / fmaxf(sqrtf(ss), 1e-12f);
  float* dst = kk + (size_t)t * CC + h * NN;
  for (int n = 0; n < NN; ++n) dst[n] = kr[n] * kkw[n] * inv;
}

// k *= (1 + (a-1)*k_a)
__global__ void fuse_k_kernel(float* __restrict__ k, const float* __restrict__ a,
                              const float* __restrict__ k_a) {
  int i = blockIdx.x * blockDim.x + threadIdx.x;
  if (i >= TT * CC) return;
  int c = i % CC;
  k[i] *= (1.f + (a[i] - 1.f) * k_a[c]);
}

// =====================================================================
// Sequential WKV scan. One 64-thread block per head; thread = value row v.
// S[v][k] kept in 64 VGPRs; per-step vectors broadcast through LDS.
//   sa = S·a ; S = S*dec + sa*b + v*k ; o = S·r   (a = -kk, b = kk*a_in)
// =====================================================================
__global__ __launch_bounds__(64)
void scan_kernel(const float* __restrict__ r, const float* __restrict__ wlog,
                 const float* __restrict__ k, const float* __restrict__ v,
                 const float* __restrict__ kk, const float* __restrict__ ain,
                 const float* __restrict__ S_in, float* __restrict__ o,
                 float* __restrict__ S_out) {
  const int h = blockIdx.x;
  const int tid = threadIdx.x;  // value index 0..63
  __shared__ float rs[NN], ds[NN], ks[NN], as[NN], bs[NN];
  float S[NN];
  const float* Sin = S_in + (size_t)h * NN * NN + (size_t)tid * NN;
#pragma unroll
  for (int j = 0; j < NN; ++j) S[j] = Sin[j];

  for (int t = 0; t < TT; ++t) {
    const int base = t * CC + h * NN;
    rs[tid] = r[base + tid];
    ds[tid] = __expf(wlog[base + tid]);
    ks[tid] = k[base + tid];
    float kv = kk[base + tid];
    as[tid] = -kv;
    bs[tid] = kv * ain[base + tid];
    float vl = v[base + tid];
    __syncthreads();
    float sa = 0.f;
#pragma unroll
    for (int j = 0; j < NN; ++j) sa += S[j] * as[j];
    float ot = 0.f;
#pragma unroll
    for (int j = 0; j < NN; ++j) {
      S[j] = S[j] * ds[j] + sa * bs[j] + vl * ks[j];
      ot += S[j] * rs[j];
    }
    o[base + tid] = ot;
    __syncthreads();
  }
  float* So = S_out + (size_t)h * NN * NN + (size_t)tid * NN;
#pragma unroll
  for (int j = 0; j < NN; ++j) So[j] = S[j];
}

// =====================================================================
// GroupNorm(H groups, eps 6.4e-4) + (r*k*r_k).sum * v term, in-place on o.
// Thread per (t,h).
// =====================================================================
__global__ void gn_out_kernel(float* __restrict__ o, const float* __restrict__ r,
                              const float* __restrict__ k, const float* __restrict__ v,
                              const float* __restrict__ r_k,
                              const float* __restrict__ gw,
                              const float* __restrict__ gb) {
  int i = blockIdx.x * blockDim.x + threadIdx.x;
  if (i >= TT * HH) return;
  int t = i / HH, h = i % HH;
  size_t base = (size_t)t * CC + h * NN;
  float m = 0.f;
  for (int n = 0; n < NN; ++n) m += o[base + n];
  m /= NN;
  float var = 0.f, dot = 0.f;
  for (int n = 0; n < NN; ++n) {
    float d = o[base + n] - m;
    var += d * d;
    dot += r[base + n] * k[base + n] * r_k[h * NN + n];
  }
  float rstd = rsqrtf(var / NN + 6.4e-4f);
  for (int n = 0; n < NN; ++n) {
    float g = (o[base + n] - m) * rstd * gw[h * NN + n] + gb[h * NN + n];
    o[base + n] = g + dot * v[base + n];
  }
}

// elementwise product (gate)
__global__ void mul_kernel(const float* __restrict__ a, const float* __restrict__ b,
                           float* __restrict__ out) {
  int i = blockIdx.x * blockDim.x + threadIdx.x;
  if (i >= TT * CC) return;
  out[i] = a[i] * b[i];
}

// =====================================================================
// Head GEMV: wave per vocab row, lane-strided dot + shfl_xor reduce (wave32)
// =====================================================================
__global__ __launch_bounds__(256)
void head_gemv(const float* __restrict__ xb, const float* __restrict__ Wv,
               float* __restrict__ out) {
  int row = blockIdx.x * 8 + (threadIdx.x >> 5);
  int lane = threadIdx.x & 31;
  if (row >= VV) return;
  float s = 0.f;
  for (int c = lane; c < CC; c += 32) s += xb[c] * Wv[(size_t)row * CC + c];
#pragma unroll
  for (int off = 16; off; off >>= 1) s += __shfl_xor(s, off, 32);
  if (lane == 0) out[row] = s;
}

// =====================================================================
// Host orchestration
// =====================================================================
extern "C" void kernel_launch(void* const* d_in, const int* in_sizes, int n_in,
                              void* d_out, int out_size, void* d_ws, size_t ws_size,
                              hipStream_t stream) {
  (void)in_sizes; (void)n_in; (void)out_size; (void)ws_size;
  const int*   idx      = (const int*)  d_in[0];
  const float* shift_in = (const float*)d_in[1];
  const float* wkv_in   = (const float*)d_in[2];
  const float* emb      = (const float*)d_in[3];
  const float* ln1_w    = (const float*)d_in[4];
  const float* ln1_b    = (const float*)d_in[5];
  const float* ln2_w    = (const float*)d_in[6];
  const float* ln2_b    = (const float*)d_in[7];
  const float* lnx_w    = (const float*)d_in[8];
  const float* lnx_b    = (const float*)d_in[9];
  const float* x_mix    = (const float*)d_in[10];
  const float* w0       = (const float*)d_in[11];
  const float* w1       = (const float*)d_in[12];
  const float* w2       = (const float*)d_in[13];
  const float* a0       = (const float*)d_in[14];
  const float* a1       = (const float*)d_in[15];
  const float* a2       = (const float*)d_in[16];
  const float* v0       = (const float*)d_in[17];
  const float* v1       = (const float*)d_in[18];
  const float* v2       = (const float*)d_in[19];
  const float* g1       = (const float*)d_in[20];
  const float* g2       = (const float*)d_in[21];
  const float* k_k      = (const float*)d_in[22];
  const float* k_a      = (const float*)d_in[23];
  const float* r_k      = (const float*)d_in[24];
  const float* Rw       = (const float*)d_in[25];
  const float* Kw       = (const float*)d_in[26];
  const float* Vw       = (const float*)d_in[27];
  const float* Ow       = (const float*)d_in[28];
  const float* ffn_xk   = (const float*)d_in[29];
  const float* ffn_K    = (const float*)d_in[30];
  const float* ffn_V    = (const float*)d_in[31];
  const float* norm_w   = (const float*)d_in[32];
  const float* norm_b   = (const float*)d_in[33];
  const float* head_w   = (const float*)d_in[34];

  float* out = (float*)d_out;
  float* out_logits = out;                       // [V]
  float* out_shifts = out + VV;                  // [L,2,1,1,C]
  float* out_wkv    = out + VV + LL * 2 * CC;    // [L,1,H,N,N]

  // workspace layout (floats); total ~137 MB
  const size_t TC = (size_t)TT * CC;
  float* ws   = (float*)d_ws;
  float* x    = ws;            // residual
  float* xln  = ws + 1 * TC;
  float* mixb = ws + 2 * TC;   // reused for xr/xk/xv/xa/xw/xg/kc/gated-xo
  float* rb   = ws + 3 * TC;
  float* kb   = ws + 4 * TC;
  float* vb   = ws + 5 * TC;
  float* wb   = ws + 6 * TC;   // wlog
  float* ab   = ws + 7 * TC;   // a (in-context learning rate)
  float* gb   = ws + 8 * TC;   // output gate
  float* vfst = ws + 9 * TC;
  float* kkb  = ws + 10 * TC;  // kk (also scratch for v-sigmoid)
  float* ob   = ws + 11 * TC;  // scan output -> xo (in place)
  float* x2   = ws + 12 * TC;
  float* ffnh = ws + 13 * TC;                       // [T,FH]
  float* tmps = ffnh + (size_t)TT * FHH;            // [T,128]
  float* lnl  = tmps + (size_t)TT * 128;            // [C]

  const dim3 blk(256);
  const dim3 gTC((TT * CC + 255) / 256);
  const dim3 gTH((TT * HH + 255) / 256);

  embed_kernel<<<gTC, blk, 0, stream>>>(idx, emb, x);

  for (int i = 0; i < LL; ++i) {
    const float* shift0 = shift_in + (size_t)(i * 2 + 0) * CC;
    const float* shift1 = shift_in + (size_t)(i * 2 + 1) * CC;

    // ---- TimeMix ----
    layernorm_kernel<<<dim3(TT), blk, 0, stream>>>(x, ln1_w + i * CC, ln1_b + i * CC, xln, 1e-5f);
    hipMemcpyAsync(out_shifts + (size_t)(i * 2 + 0) * CC, xln + (size_t)(TT - 1) * CC,
                   CC * sizeof(float), hipMemcpyDeviceToDevice, stream);

    // r = xr @ Rw^T
    mix_kernel<<<gTC, blk, 0, stream>>>(xln, shift0, x_mix + (size_t)(i * 6 + 0) * CC, mixb);
    gemm_nt_wmma<false, false><<<dim3(TT / 32, CC / 128), blk, 0, stream>>>(
        mixb, Rw + (size_t)i * CC * CC, rb, TT, CC, CC);
    // k = xk @ Kw^T
    mix_kernel<<<gTC, blk, 0, stream>>>(xln, shift0, x_mix + (size_t)(i * 6 + 2) * CC, mixb);
    gemm_nt_wmma<false, false><<<dim3(TT / 32, CC / 128), blk, 0, stream>>>(
        mixb, Kw + (size_t)i * CC * CC, kb, TT, CC, CC);
    // v = xv @ Vw^T  (keep xv in mixb for v-LoRA)
    mix_kernel<<<gTC, blk, 0, stream>>>(xln, shift0, x_mix + (size_t)(i * 6 + 3) * CC, mixb);
    gemm_nt_wmma<false, false><<<dim3(TT / 32, CC / 128), blk, 0, stream>>>(
        mixb, Vw + (size_t)i * CC * CC, vb, TT, CC, CC);
    if (i == 0) {
      hipMemcpyAsync(vfst, vb, TC * sizeof(float), hipMemcpyDeviceToDevice, stream);
    } else {
      small_gemm_nn<<<dim3((TT * DVV + 255) / 256), blk, 0, stream>>>(
          mixb, v1 + (size_t)i * CC * DVV, tmps, CC, DVV, 0);
      lora_out_kernel<<<gTC, blk, 0, stream>>>(tmps, v2 + (size_t)i * DVV * CC,
                                               v0 + i * CC, kkb, DVV, 1);  // sigmoid -> kkb scratch
      vblend_kernel<<<gTC, blk, 0, stream>>>(vb, vfst, kkb);
    }
    // a = sigmoid(xa@a1@a2 + a0)
    mix_kernel<<<gTC, blk, 0, stream>>>(xln, shift0, x_mix + (size_t)(i * 6 + 4) * CC, mixb);
    small_gemm_nn<<<dim3((TT * DAA + 255) / 256), blk, 0, stream>>>(
        mixb, a1 + (size_t)i * CC * DAA, tmps, CC, DAA, 0);
    lora_out_kernel<<<gTC, blk, 0, stream>>>(tmps, a2 + (size_t)i * DAA * CC,
                                             a0 + i * CC, ab, DAA, 1);
    // kk from pre-fused k, then fuse k
    kk_norm_kernel<<<gTH, blk, 0, stream>>>(kb, k_k + i * CC, kkb);
    fuse_k_kernel<<<gTC, blk, 0, stream>>>(kb, ab, k_a + i * CC);
    // wlog = -exp(-softplus(-(tanh(xw@w1)@w2 + w0)) - 0.5)
    mix_kernel<<<gTC, blk, 0, stream>>>(xln, shift0, x_mix + (size_t)(i * 6 + 1) * CC, mixb);
    small_gemm_nn<<<dim3((TT * DWW + 255) / 256), blk, 0, stream>>>(
        mixb, w1 + (size_t)i * CC * DWW, tmps, CC, DWW, 1);
    lora_out_kernel<<<gTC, blk, 0, stream>>>(tmps, w2 + (size_t)i * DWW * CC,
                                             w0 + i * CC, wb, DWW, 2);
    // gate = sigmoid(xg@g1)@g2
    mix_kernel<<<gTC, blk, 0, stream>>>(xln, shift0, x_mix + (size_t)(i * 6 + 5) * CC, mixb);
    small_gemm_nn<<<dim3((TT * DGG + 255) / 256), blk, 0, stream>>>(
        mixb, g1 + (size_t)i * CC * DGG, tmps, CC, DGG, 2);
    lora_out_kernel<<<gTC, blk, 0, stream>>>(tmps, g2 + (size_t)i * DGG * CC,
                                             nullptr, gb, DGG, 0);

    // sequential WKV scan; final state straight into d_out
    scan_kernel<<<dim3(HH), dim3(64), 0, stream>>>(
        rb, wb, kb, vb, kkb, ab,
        wkv_in + (size_t)i * HH * NN * NN, ob,
        out_wkv + (size_t)i * HH * NN * NN);

    // group-norm + bonus term (in-place on ob), gate, O-proj with residual
    gn_out_kernel<<<gTH, blk, 0, stream>>>(ob, rb, kb, vb, r_k + i * HH * NN,
                                           lnx_w + i * CC, lnx_b + i * CC);
    mul_kernel<<<gTC, blk, 0, stream>>>(ob, gb, mixb);
    gemm_nt_wmma<true, false><<<dim3(TT / 32, CC / 128), blk, 0, stream>>>(
        mixb, Ow + (size_t)i * CC * CC, x, TT, CC, CC);

    // ---- ChannelMix ----
    layernorm_kernel<<<dim3(TT), blk, 0, stream>>>(x, ln2_w + i * CC, ln2_b + i * CC, x2, 1e-5f);
    hipMemcpyAsync(out_shifts + (size_t)(i * 2 + 1) * CC, x2 + (size_t)(TT - 1) * CC,
                   CC * sizeof(float), hipMemcpyDeviceToDevice, stream);
    mix_kernel<<<gTC, blk, 0, stream>>>(x2, shift1, ffn_xk + i * CC, mixb);
    gemm_nt_wmma<false, true><<<dim3(TT / 32, FHH / 128), blk, 0, stream>>>(
        mixb, ffn_K + (size_t)i * FHH * CC, ffnh, TT, FHH, CC);   // relu^2 fused
    gemm_nt_wmma<true, false><<<dim3(TT / 32, CC / 128), blk, 0, stream>>>(
        ffnh, ffn_V + (size_t)i * CC * FHH, x, TT, CC, FHH);      // residual acc
  }

  // final LN on last token + head
  layernorm_kernel<<<dim3(1), blk, 0, stream>>>(x + (size_t)(TT - 1) * CC,
                                                norm_w, norm_b, lnl, 1e-5f);
  head_gemv<<<dim3((VV + 7) / 8), blk, 0, stream>>>(lnl, head_w, out_logits);
}
